// EncoderModule_30073361006812
// MI455X (gfx1250) — matrix operand order repaired and checked by reading, not compiled, not run on hardware
//
#include <hip/hip_runtime.h>
#include <math.h>

#define DEV __device__ __forceinline__

typedef __attribute__((ext_vector_type(16))) __bf16 v16bf;
typedef __attribute__((ext_vector_type(8)))  float  v8f;

// ---------- helpers ----------
DEV unsigned short f2bf(float x) {           // f32 -> bf16 (RNE)
  unsigned u = __float_as_uint(x);
  u += 0x7FFFu + ((u >> 16) & 1u);
  return (unsigned short)(u >> 16);
}
DEV float gelu_exact(float x) { return 0.5f * x * (1.0f + erff(x * 0.7071067811865476f)); }
DEV int   f2ord(float f) { int i = __float_as_int(f); return i >= 0 ? i : (i ^ 0x7FFFFFFF); }
DEV float ord2f(int i)   { return __int_as_float(i >= 0 ? i : (i ^ 0x7FFFFFFF)); }

// CDNA5 async global->LDS copy (ASYNCcnt path). lds_off = LDS byte offset
// (low 32 bits of generic pointer == LDS offset per aperture rules).
DEV void async_copy16(unsigned lds_off, unsigned long long gaddr) {
  asm volatile("global_load_async_to_lds_b128 %0, %1, off"
               :: "v"(lds_off), "v"(gaddr) : "memory");
}
DEV void wait_async() { asm volatile("s_wait_asynccnt 0" ::: "memory"); }
DEV unsigned lds_off_of(const void* p) { return (unsigned)(unsigned long long)p; }

union BFrag { v16bf v; unsigned u[8]; };

DEV v8f wmma_bf16(const unsigned a[8], const unsigned b[8], v8f c) {
  BFrag A, B;
#pragma unroll
  for (int i = 0; i < 8; ++i) { A.u[i] = a[i]; B.u[i] = b[i]; }
  return __builtin_amdgcn_wmma_f32_16x16x32_bf16(false, A.v, false, B.v, (short)0, c, false, false);
}

// A fragment: 16x32 bf16. lanes 0-15 rows 0-15 hold K {0..7,16..23}; lanes 16-31 same rows, K {8..15,24..31}
DEV void load_a_frag(const unsigned short* A, int stride, int row, int hf, int kc, unsigned a[8]) {
  const unsigned short* p = A + row * stride + kc * 32 + 8 * hf;
#pragma unroll
  for (int j = 0; j < 8; ++j) {
    int k = (j < 4) ? (2 * j) : (16 + 2 * (j - 4));
    a[j] = *(const unsigned*)(p + k);
  }
}
// B fragment: 32x16 bf16 from W[out][K] (B[k][n] = W[colbase+n][k]).
DEV void load_b_frag(const unsigned short* W, int stride, int colbase, int lm, int hf, int kc, unsigned b[8]) {
  const unsigned short* p = W + (colbase + lm) * stride + kc * 32 + 8 * hf;
#pragma unroll
  for (int j = 0; j < 8; ++j) b[j] = *(const unsigned*)(p + ((j < 4) ? (2 * j) : (16 + 2 * (j - 4))));
}

// ---------- per-wave 16x128 GEMM pieces (Nout = 128, 8 N-tiles) ----------
DEV void wave_gemm_to_lds(const unsigned short* A, int astride, const unsigned short* Wl,
                          int kchunks, const float* biasl, unsigned short* Out,
                          int row0, int lane, int do_gelu) {
  int hf = lane >> 4, lm = lane & 15;
  for (int nt = 0; nt < 8; ++nt) {
    v8f acc = {};
    for (int kc = 0; kc < kchunks; ++kc) {
      unsigned a[8], b[8];
      load_a_frag(A, astride, row0 + lm, hf, kc, a);
      load_b_frag(Wl, astride, nt * 16, lm, hf, kc, b);
      acc = wmma_bf16(a, b, acc);
    }
    float bc = biasl[nt * 16 + lm];
#pragma unroll
    for (int r = 0; r < 8; ++r) {
      float y = acc[r] + bc;
      if (do_gelu) y = gelu_exact(y);
      Out[(row0 + r + 8 * hf) * 128 + nt * 16 + lm] = f2bf(y);
    }
  }
}

DEV void wave_gemm_to_lds_f32(const unsigned short* A, int astride, const unsigned short* Wl,
                              int kchunks, const float* biasl, float* Out, int ostride,
                              int row0, int lane) {
  int hf = lane >> 4, lm = lane & 15;
  for (int nt = 0; nt < 8; ++nt) {
    v8f acc = {};
    for (int kc = 0; kc < kchunks; ++kc) {
      unsigned a[8], b[8];
      load_a_frag(A, astride, row0 + lm, hf, kc, a);
      load_b_frag(Wl, astride, nt * 16, lm, hf, kc, b);
      acc = wmma_bf16(a, b, acc);
    }
    float bc = biasl[nt * 16 + lm];
#pragma unroll
    for (int r = 0; r < 8; ++r)
      Out[(row0 + r + 8 * hf) * ostride + nt * 16 + lm] = acc[r] + bc;
  }
}

DEV void wave_gemm_to_global(const unsigned short* A, int astride, const unsigned short* Wl,
                             int kchunks, const float* biasl, float* __restrict__ Y,
                             int e0, int E, int row0, int lane) {
  int hf = lane >> 4, lm = lane & 15;
  for (int nt = 0; nt < 8; ++nt) {
    v8f acc = {};
    for (int kc = 0; kc < kchunks; ++kc) {
      unsigned a[8], b[8];
      load_a_frag(A, astride, row0 + lm, hf, kc, a);
      load_b_frag(Wl, astride, nt * 16, lm, hf, kc, b);
      acc = wmma_bf16(a, b, acc);
    }
    float bc = biasl[nt * 16 + lm];
#pragma unroll
    for (int r = 0; r < 8; ++r) {
      int er = e0 + row0 + r + 8 * hf;
      if (er < E) Y[(size_t)er * 128 + nt * 16 + lm] = acc[r] + bc;
    }
  }
}

// logits[e,k] = leaky_relu(h @ aW_k^T + b_k) . aA_k + aAb_k
DEV void wave_head_logits(const unsigned short* A, int astride, const unsigned short* Wl,
                          int kchunks, const float* biasl, const float* aAl, float aAb,
                          float* __restrict__ logits, int e0, int E, int k,
                          int row0, int lane) {
  int hf = lane >> 4, lm = lane & 15;
  float part[8] = {0.f, 0.f, 0.f, 0.f, 0.f, 0.f, 0.f, 0.f};
  for (int nt = 0; nt < 8; ++nt) {
    v8f acc = {};
    for (int kc = 0; kc < kchunks; ++kc) {
      unsigned a[8], b[8];
      load_a_frag(A, astride, row0 + lm, hf, kc, a);
      load_b_frag(Wl, astride, nt * 16, lm, hf, kc, b);
      acc = wmma_bf16(a, b, acc);
    }
    float bc = biasl[nt * 16 + lm];
    float av = aAl[nt * 16 + lm];
#pragma unroll
    for (int r = 0; r < 8; ++r) {
      float y = acc[r] + bc;
      y = (y > 0.f) ? y : 0.2f * y;   // leaky_relu 0.2
      part[r] += y * av;
    }
  }
#pragma unroll
  for (int off = 8; off > 0; off >>= 1)
#pragma unroll
    for (int r = 0; r < 8; ++r) part[r] += __shfl_xor(part[r], off);
  if (lm == 0) {
#pragma unroll
    for (int r = 0; r < 8; ++r) {
      int er = e0 + row0 + r + 8 * hf;
      if (er < E) logits[(size_t)er * 4 + k] = part[r] + aAb;
    }
  }
}

// ---------- edge kernel: mode 0 = msg MLP + head logits; mode 1 = edge-update MLP + LN ----------
// smem (bytes): src[128]i32 @0, snk[128]i32 @512, bias[128]f32 @1024, aA[128]f32 @1536,
//   h bf16[128*384] @2048, w bf16[128*384] @100352, t1 bf16[128*128] @198656, t2 @231424
#define SM_EDGE 264192
__global__ __launch_bounds__(256) void edge_mlp_kernel(
    const unsigned short* __restrict__ nodes16,   // [*,128] bf16 (pre-converted)
    const float* __restrict__ eattr,
    const int* __restrict__ eidx, int E,
    const unsigned short* __restrict__ W1, const float* __restrict__ b1,   // bf16 [128,384]
    const unsigned short* __restrict__ W2, const float* __restrict__ b2,   // bf16 [128,128]
    const unsigned short* __restrict__ W3, const float* __restrict__ b3,   // bf16 [128,128]
    float* __restrict__ msg_out,
    const unsigned short* __restrict__ aW16, const float* __restrict__ aWb,
    const float* __restrict__ aAw, const float* __restrict__ aAb,
    float* __restrict__ logits_out,
    const float* __restrict__ lng, const float* __restrict__ lnb,
    float* __restrict__ eout, int mode) {
  extern __shared__ unsigned char smem[];
  int* s_src = (int*)smem;
  int* s_snk = (int*)(smem + 512);
  float* s_bias = (float*)(smem + 1024);
  float* s_aA = (float*)(smem + 1536);
  unsigned short* s_h  = (unsigned short*)(smem + 2048);
  unsigned short* s_w  = (unsigned short*)(smem + 100352);
  unsigned short* s_t1 = (unsigned short*)(smem + 198656);
  unsigned short* s_t2 = (unsigned short*)(smem + 231424);
  float* s_ln = (float*)(smem + 2048);   // mode 1 only: overlays h (dead after layer 1), stride 129

  const int tid = threadIdx.x, lane = tid & 31, wid = tid >> 5;
  const int e0 = blockIdx.x * 128;
  const unsigned sh_off = lds_off_of(s_h);
  const unsigned sw_off = lds_off_of(s_w);

  for (int i = tid; i < 128; i += 256) {
    int e = e0 + i, s = 0, d = 0;
    if (e < E) { s = eidx[e]; d = eidx[E + e]; }
    s_src[i] = s; s_snk[i] = d;
  }
  __syncthreads();
  // async: W1 (98304 B) and node halves of h (bf16 gather rows, 256 B each)
  for (int t = tid; t < 6144; t += 256)
    async_copy16(sw_off + t * 16, (unsigned long long)((const char*)W1 + t * 16));
  for (int t = tid; t < 4096; t += 256) {
    int row = t >> 5, seg = (t >> 4) & 1, ch = t & 15;
    int nidx = seg ? s_snk[row] : s_src[row];
    async_copy16(sh_off + row * 768 + seg * 256 + ch * 16,
                 (unsigned long long)((const char*)nodes16 + (size_t)nidx * 256 + ch * 16));
  }
  // eattr third of h: f32 -> bf16 on VALU
  for (int t = tid; t < 128 * 128; t += 256) {
    int row = t >> 7, c = t & 127;
    int e = e0 + row;
    float v = (e < E) ? eattr[(size_t)e * 128 + c] : 0.f;
    s_h[row * 384 + 256 + c] = f2bf(v);
  }
  if (tid < 128) s_bias[tid] = b1[tid];
  wait_async();
  __syncthreads();
  wave_gemm_to_lds(s_h, 384, s_w, 12, s_bias, s_t1, wid * 16, lane, 1);
  __syncthreads();
  for (int t = tid; t < 2048; t += 256)
    async_copy16(sw_off + t * 16, (unsigned long long)((const char*)W2 + t * 16));
  if (tid < 128) s_bias[tid] = b2[tid];
  wait_async();
  __syncthreads();
  wave_gemm_to_lds(s_t1, 128, s_w, 4, s_bias, s_t2, wid * 16, lane, 1);
  __syncthreads();
  for (int t = tid; t < 2048; t += 256)
    async_copy16(sw_off + t * 16, (unsigned long long)((const char*)W3 + t * 16));
  if (tid < 128) s_bias[tid] = b3[tid];
  wait_async();
  __syncthreads();

  if (mode == 0) {
    wave_gemm_to_global(s_t2, 128, s_w, 4, s_bias, msg_out, e0, E, wid * 16, lane);
    for (int k = 0; k < 4; ++k) {
      __syncthreads();
      for (int t = tid; t < 6144; t += 256)
        async_copy16(sw_off + t * 16,
                     (unsigned long long)((const char*)aW16 + (size_t)k * 98304 + t * 16));
      if (tid < 128) { s_bias[tid] = aWb[k * 128 + tid]; s_aA[tid] = aAw[k * 128 + tid]; }
      wait_async();
      __syncthreads();
      wave_head_logits(s_h, 384, s_w, 12, s_bias, s_aA, aAb[k], logits_out, e0, E, k, wid * 16, lane);
    }
  } else {
    // eu3 (f32) -> LDS, then eout = LN(eattr + eu3)
    wave_gemm_to_lds_f32(s_t2, 128, s_w, 4, s_bias, s_ln, 129, wid * 16, lane);
    __syncthreads();
    for (int row = wid; row < 128; row += 8) {   // one wave per row (uniform)
      int e = e0 + row;
      if (e >= E) continue;
      const float* ea = eattr + (size_t)e * 128;
      float v[4];
#pragma unroll
      for (int i = 0; i < 4; ++i) v[i] = s_ln[row * 129 + lane * 4 + i] + ea[lane * 4 + i];
      float s = v[0] + v[1] + v[2] + v[3];
#pragma unroll
      for (int off = 16; off > 0; off >>= 1) s += __shfl_xor(s, off);
      float mu = s * (1.f / 128.f);
      float q = 0.f;
#pragma unroll
      for (int i = 0; i < 4; ++i) { float d = v[i] - mu; q += d * d; }
#pragma unroll
      for (int off = 16; off > 0; off >>= 1) q += __shfl_xor(q, off);
      float inv = rsqrtf(q * (1.f / 128.f) + 1e-5f);
#pragma unroll
      for (int i = 0; i < 4; ++i) {
        int j = lane * 4 + i;
        eout[(size_t)e * 128 + j] = (v[i] - mu) * inv * lng[j] + lnb[j];
      }
    }
  }
}

// ---------- generic tiled GEMM: Y[R,Nout] = act(X[R,K] @ W16[Nout,K]^T + bias) ----------
__global__ __launch_bounds__(256) void gemm_kernel(
    const float* __restrict__ X, const unsigned short* __restrict__ W16,
    const float* __restrict__ bias, float* __restrict__ Y,
    int R, int K, int Nout, int act) {
  extern __shared__ unsigned char smem[];
  float* s_bias = (float*)smem;
  unsigned short* s_x = (unsigned short*)(smem + ((Nout * 4 + 255) & ~255));
  unsigned short* s_w = s_x + 128 * K;
  const int tid = threadIdx.x, lane = tid & 31, wid = tid >> 5;
  const int r0 = blockIdx.x * 128;
  const unsigned sw_off = lds_off_of(s_w);
  for (int t = tid; t < (Nout * K) / 8; t += 256)   // async bf16 weight stage
    async_copy16(sw_off + t * 16, (unsigned long long)((const char*)W16 + t * 16));
  for (int t = tid; t < Nout; t += 256) s_bias[t] = bias[t];
  for (int t = tid; t < 128 * K; t += 256) {
    int row = t / K, c = t - row * K;
    int gr = r0 + row; if (gr >= R) gr = R - 1;
    s_x[t] = f2bf(X[(size_t)gr * K + c]);
  }
  wait_async();
  __syncthreads();
  const int hf = lane >> 4, lm = lane & 15, row0 = wid * 16;
  const int kch = K >> 5, ntl = Nout >> 4;
  for (int nt = 0; nt < ntl; ++nt) {
    v8f acc = {};
    for (int kc = 0; kc < kch; ++kc) {
      unsigned a[8], b[8];
      load_a_frag(s_x, K, row0 + lm, hf, kc, a);
      load_b_frag(s_w, K, nt * 16, lm, hf, kc, b);
      acc = wmma_bf16(a, b, acc);
    }
    float bc = s_bias[nt * 16 + lm];
#pragma unroll
    for (int r = 0; r < 8; ++r) {
      int gr = r0 + row0 + r + 8 * hf;
      if (gr < R) {
        float y = acc[r] + bc;
        if (act == 1) y = gelu_exact(y);
        Y[(size_t)gr * Nout + nt * 16 + lm] = y;
      }
    }
  }
}

// ---------- LayerNorm(a + b) over 128 dims, one wave per row ----------
__global__ __launch_bounds__(256) void ln_res_kernel(
    const float* __restrict__ A, const float* __restrict__ B,
    const float* __restrict__ g, const float* __restrict__ bt,
    float* __restrict__ out, int R) {
  int wid = threadIdx.x >> 5, lane = threadIdx.x & 31;
  int row = blockIdx.x * 8 + wid;
  if (row >= R) return;
  float4 va = ((const float4*)(A + (size_t)row * 128))[lane];
  float4 vb = ((const float4*)(B + (size_t)row * 128))[lane];
  float v[4] = {va.x + vb.x, va.y + vb.y, va.z + vb.z, va.w + vb.w};
  float s = v[0] + v[1] + v[2] + v[3];
#pragma unroll
  for (int off = 16; off > 0; off >>= 1) s += __shfl_xor(s, off);
  float mu = s * (1.f / 128.f);
  float q = 0.f;
#pragma unroll
  for (int i = 0; i < 4; ++i) { float d = v[i] - mu; q += d * d; }
#pragma unroll
  for (int off = 16; off > 0; off >>= 1) q += __shfl_xor(q, off);
  float inv = rsqrtf(q * (1.f / 128.f) + 1e-5f);
  float4 g4 = ((const float4*)g)[lane], b4 = ((const float4*)bt)[lane];
  float4 o;
  o.x = (v[0] - mu) * inv * g4.x + b4.x;
  o.y = (v[1] - mu) * inv * g4.y + b4.y;
  o.z = (v[2] - mu) * inv * g4.z + b4.z;
  o.w = (v[3] - mu) * inv * g4.w + b4.w;
  ((float4*)(out + (size_t)row * 128))[lane] = o;
}

// ---------- small kernels ----------
__global__ void cvt_bf16_kernel(const float* __restrict__ src, unsigned short* __restrict__ dst, int n) {
  int t = blockIdx.x * blockDim.x + threadIdx.x;
  if (t < n) dst[t] = f2bf(src[t]);
}
__global__ void init_kernel(int* __restrict__ smaxi, float* __restrict__ norm,
                            float* __restrict__ agg, int N) {
  int t = blockIdx.x * blockDim.x + threadIdx.x;
  if (t < N * 512) agg[t] = 0.f;
  if (t < N * 4) { smaxi[t] = (int)0x80000000; norm[t] = 0.f; }
}
__global__ void seg_max_kernel(const float* __restrict__ logits, const int* __restrict__ snk,
                               int* __restrict__ smaxi, int E) {
  int t = blockIdx.x * blockDim.x + threadIdx.x;
  if (t >= E * 4) return;
  int e = t >> 2, k = t & 3;
  atomicMax(&smaxi[snk[e] * 4 + k], f2ord(logits[t]));
}
__global__ void att_exp_kernel(const float* __restrict__ logits, const int* __restrict__ snk,
                               const int* __restrict__ smaxi, float* __restrict__ attexp,
                               float* __restrict__ norm, int E) {
  int t = blockIdx.x * blockDim.x + threadIdx.x;
  if (t >= E * 4) return;
  int e = t >> 2, k = t & 3;
  int s = snk[e];
  float a = expf(logits[t] - ord2f(smaxi[s * 4 + k]));
  attexp[t] = a;
  atomicAdd(&norm[s * 4 + k], a + 1e-12f);
}
__global__ void att_norm_kernel(float* __restrict__ att, const int* __restrict__ snk,
                                const float* __restrict__ norm, int E) {
  int t = blockIdx.x * blockDim.x + threadIdx.x;
  if (t >= E * 4) return;
  int e = t >> 2, k = t & 3;
  att[t] = att[t] / norm[snk[e] * 4 + k];
}
// agg[snk[e], k, :] += att[snk[e], k] * msg[e, :]   (faithful att[snk] re-gather quirk)
__global__ __launch_bounds__(128) void aggregate_kernel(
    const float* __restrict__ msg, const float* __restrict__ att,
    const int* __restrict__ snk, float* __restrict__ agg, int E) {
  int e = blockIdx.x, d = threadIdx.x;
  int s = snk[e];
  float m = msg[(size_t)e * 128 + d];
#pragma unroll
  for (int k = 0; k < 4; ++k) {
    float w = att[s * 4 + k];
    atomicAdd(&agg[(size_t)s * 512 + k * 128 + d], w * m);
  }
}

// ---------- host ----------
extern "C" void kernel_launch(void* const* d_in, const int* in_sizes, int n_in,
                              void* d_out, int out_size, void* d_ws, size_t ws_size,
                              hipStream_t stream) {
  (void)n_in; (void)out_size; (void)ws_size;
  const float* bb     = (const float*)d_in[0];
  const float* eattr  = (const float*)d_in[1];
  const int*   eidx   = (const int*)d_in[2];
  const float* aWw    = (const float*)d_in[3];
  const float* aWb    = (const float*)d_in[4];
  const float* aAw    = (const float*)d_in[5];
  const float* aAb    = (const float*)d_in[6];
  const float* nw1    = (const float*)d_in[7];
  const float* nb1    = (const float*)d_in[8];
  const float* nw2    = (const float*)d_in[9];
  const float* nb2    = (const float*)d_in[10];
  const float* nw3    = (const float*)d_in[11];
  const float* nb3    = (const float*)d_in[12];
  const float* dw1    = (const float*)d_in[13];
  const float* db1    = (const float*)d_in[14];
  const float* dw2    = (const float*)d_in[15];
  const float* db2    = (const float*)d_in[16];
  const float* ew1    = (const float*)d_in[17];
  const float* eb1    = (const float*)d_in[18];
  const float* ew2    = (const float*)d_in[19];
  const float* eb2    = (const float*)d_in[20];
  const float* ew3    = (const float*)d_in[21];
  const float* eb3    = (const float*)d_in[22];
  const float* aggrw  = (const float*)d_in[23];
  const float* aggrb  = (const float*)d_in[24];
  const float* ln1g   = (const float*)d_in[25];
  const float* ln1b   = (const float*)d_in[26];
  const float* lneg   = (const float*)d_in[27];
  const float* lneb   = (const float*)d_in[28];

  const int N = in_sizes[0] / 128;
  const int E = in_sizes[1] / 128;
  const int* snk = eidx + E;

  // workspace carve (f32). msg region [0, E*128) is dead after aggregation and
  // is re-used for the node-phase buffers (N*896 floats << E*128).
  float* ws     = (float*)d_ws;
  float* msg    = ws;
  float* upd    = ws;
  float* xbuf   = upd + (size_t)N * 128;
  float* tbuf   = xbuf + (size_t)N * 128;
  float* dense  = tbuf + (size_t)N * 512;
  float* logits = ws + (size_t)E * 128;
  float* attexp = logits + (size_t)E * 4;
  int*   smaxi  = (int*)(attexp + (size_t)E * 4);
  float* norm   = (float*)(smaxi + (size_t)N * 4);
  float* agg    = norm + (size_t)N * 4;
  // bf16 staging area (16B-aligned): pre-converted weights + node feature tables
  unsigned short* nw1_16 = (unsigned short*)(agg + (size_t)N * 512);
  unsigned short* nw2_16 = nw1_16 + 49152;
  unsigned short* nw3_16 = nw2_16 + 16384;
  unsigned short* aW16   = nw3_16 + 16384;          // 4*128*384
  unsigned short* ew1_16 = aW16 + 196608;
  unsigned short* ew2_16 = ew1_16 + 49152;
  unsigned short* ew3_16 = ew2_16 + 16384;
  unsigned short* agw16  = ew3_16 + 16384;          // 128*512
  unsigned short* dw1_16 = agw16 + 65536;           // 512*128
  unsigned short* dw2_16 = dw1_16 + 65536;          // 128*512
  unsigned short* bb16   = dw2_16 + 65536;          // N*128
  unsigned short* x16    = bb16 + (size_t)N * 128;  // N*128

  float* x_out = (float*)d_out;                     // [N,128]
  float* e_out = (float*)d_out + (size_t)N * 128;   // [E,128]

  const int ebl = (E + 127) / 128;
  auto smem_gemm = [](int K, int Nout) -> size_t {
    return (size_t)((Nout * 4 + 255) & ~255) + (size_t)(128 * K + Nout * K) * 2;
  };
  auto cvt = [&](const float* s, unsigned short* d, int n) {
    cvt_bf16_kernel<<<(n + 255) / 256, 256, 0, stream>>>(s, d, n);
  };

  // pre-convert weights / node features to bf16 (once per launch, deterministic)
  cvt(nw1, nw1_16, 49152);  cvt(nw2, nw2_16, 16384);  cvt(nw3, nw3_16, 16384);
  cvt(aWw, aW16, 196608);
  cvt(ew1, ew1_16, 49152);  cvt(ew2, ew2_16, 16384);  cvt(ew3, ew3_16, 16384);
  cvt(aggrw, agw16, 65536); cvt(dw1, dw1_16, 65536);  cvt(dw2, dw2_16, 65536);
  cvt(bb, bb16, N * 128);

  init_kernel<<<(N * 512 + 255) / 256, 256, 0, stream>>>(smaxi, norm, agg, N);

  // edge message MLP + per-head attention logits (WMMA bf16, async LDS staging)
  edge_mlp_kernel<<<ebl, 256, SM_EDGE, stream>>>(
      bb16, eattr, eidx, E, nw1_16, nb1, nw2_16, nb2, nw3_16, nb3,
      msg, aW16, aWb, aAw, aAb, logits, nullptr, nullptr, nullptr, 0);

  seg_max_kernel<<<(E * 4 + 255) / 256, 256, 0, stream>>>(logits, snk, smaxi, E);
  att_exp_kernel<<<(E * 4 + 255) / 256, 256, 0, stream>>>(logits, snk, smaxi, attexp, norm, E);
  att_norm_kernel<<<(E * 4 + 255) / 256, 256, 0, stream>>>(attexp, snk, norm, E);
  aggregate_kernel<<<E, 128, 0, stream>>>(msg, attexp, snk, agg, E);

  // node pipeline
  gemm_kernel<<<(N + 127) / 128, 256, smem_gemm(512, 128), stream>>>(agg, agw16, aggrb, upd, N, 512, 128, 0);
  ln_res_kernel<<<(N + 7) / 8, 256, 0, stream>>>(bb, upd, ln1g, ln1b, xbuf, N);
  gemm_kernel<<<(N + 127) / 128, 256, smem_gemm(128, 512), stream>>>(xbuf, dw1_16, db1, tbuf, N, 128, 512, 1);
  gemm_kernel<<<(N + 127) / 128, 256, smem_gemm(512, 128), stream>>>(tbuf, dw2_16, db2, dense, N, 512, 128, 0);
  ln_res_kernel<<<(N + 7) / 8, 256, 0, stream>>>(dense, upd, ln1g, ln1b, x_out, N);  // ln1 reused (faithful)

  // edge-update MLP + LN (x gathered from bf16 copy of x_out)
  cvt(x_out, x16, N * 128);
  edge_mlp_kernel<<<ebl, 256, SM_EDGE, stream>>>(
      x16, eattr, eidx, E, ew1_16, eb1, ew2_16, eb2, ew3_16, eb3,
      nullptr, nullptr, nullptr, nullptr, nullptr, nullptr, lneg, lneb, e_out, 1);
}
